// MultiHeadAttention_39195871544049
// MI455X (gfx1250) — matrix operand order repaired and checked by reading, not compiled, not run on hardware
//
#include <hip/hip_runtime.h>
#include <hip/hip_bf16.h>

// MHA: B=2, T=2048, C=1024, H=16, hd=64. All matrix math on f16 WMMA (f32 acc).
// Compute-bound (~69 GFLOP vs ~50MB unique bytes); working set fits in 192MB L2.

typedef __attribute__((ext_vector_type(16))) _Float16 v16h;
typedef __attribute__((ext_vector_type(8)))  _Float16 v8h;
typedef __attribute__((ext_vector_type(8)))  float    v8f;
typedef __attribute__((ext_vector_type(4)))  unsigned int v4u;
typedef __attribute__((ext_vector_type(4)))  int v4i;
typedef __attribute__((ext_vector_type(8)))  int v8i;

union V16HU { v16h v; v8h h[2]; };

#define T_SEQ 2048
#define C_DIM 1024
#define NEG_BIG (-1e30f)

#if defined(__has_builtin)
#if __has_builtin(__builtin_amdgcn_tensor_load_to_lds)
#define HAVE_TDM 1
#endif
#endif
#ifndef HAVE_TDM
#define HAVE_TDM 0
#warning "TDM builtin unavailable: attention falls back to direct global V loads"
#endif

// ---- WMMA fragment helpers (layouts per CDNA5 ISA 7.12.2) ----

// A fragment, 16x32 f16, row-major memory [16 x K], ld = row stride (elems).
// lane<16: row=lane, K={0..7,16..23}; lane>=16: row=lane-16, K={8..15,24..31}
__device__ __forceinline__ v16h load_a_f16(const _Float16* __restrict__ p, int ld, int lane) {
  const _Float16* q = p + (long)(lane & 15) * ld + ((lane >> 4) << 3);
  V16HU u;
  u.h[0] = *(const v8h*)(q);
  u.h[1] = *(const v8h*)(q + 16);
  return u.v;
}

// B fragment, 32x16 f16 (KxN), from COLUMN-major memory, i.e. memory is
// [N x K] row-major with row stride ld. lane<16: col=lane, K={0..7,16..23};
// lane>=16: col=lane-16, K={8..15,24..31} (same K order as A fragment).
__device__ __forceinline__ v16h load_b_f16(const _Float16* __restrict__ p, int ld, int lane) {
  const _Float16* q = p + (long)(lane & 15) * ld + ((lane >> 4) << 3);
  V16HU u;
  u.h[0] = *(const v8h*)(q);
  u.h[1] = *(const v8h*)(q + 16);
  return u.v;
}

__device__ __forceinline__ v8f wmma_f32(v16h a, v16h b, v8f c) {
  return __builtin_amdgcn_wmma_f32_16x16x32_f16(false, a, false, b, (short)0, c, false, false);
}

#if HAVE_TDM
// TDM: DMA a 2-D tile (tile_d0 cols x tile_d1 rows, 2-byte elems) from a
// row-major tensor into LDS. gaddr is the TILE start; tensor_d0 is the
// remaining extent from the tile start so OOB columns zero-fill.
// D# packing per CDNA5 ISA.
__device__ __forceinline__ void tdm_load_2d_f16(unsigned lds_off, unsigned long gaddr,
                                                unsigned tile_d0, unsigned tile_d1,
                                                unsigned tensor_d0, unsigned tensor_d1,
                                                unsigned long stride0) {
  v4u g0;
  g0[0] = 1u;                                            // count=1 (valid), user mode
  g0[1] = lds_off;                                       // lds_addr (bytes)
  g0[2] = (unsigned)(gaddr & 0xffffffffu);               // global_addr[31:0]
  g0[3] = (unsigned)((gaddr >> 32) & 0x01ffffffu) | (2u << 30);  // addr[56:32] | type=2
  v8i g1;
  g1[0] = (int)(1u << 16);                               // data_size=1 (2 bytes)
  g1[1] = (int)((tensor_d0 & 0xffffu) << 16);            // tensor_dim0[15:0]
  g1[2] = (int)((tensor_d0 >> 16) | ((tensor_d1 & 0xffffu) << 16));
  g1[3] = (int)((tensor_d1 >> 16) | ((tile_d0 & 0xffffu) << 16));   // tile_dim0
  g1[4] = (int)(tile_d1 & 0xffffu);                      // tile_dim1 (tile_dim2=0)
  g1[5] = (int)(unsigned)(stride0 & 0xffffffffu);        // tensor_dim0_stride[31:0]
  g1[6] = (int)(unsigned)((stride0 >> 32) & 0xffffu);    // stride[47:32]
  g1[7] = 0;
  v4i z = {0, 0, 0, 0};
#if __clang_major__ >= 23
  v8i z8 = {0, 0, 0, 0, 0, 0, 0, 0};
  __builtin_amdgcn_tensor_load_to_lds(g0, g1, z, z, z8, 0);
#else
  __builtin_amdgcn_tensor_load_to_lds(g0, g1, z, z, 0);
#endif
}
#endif

// ---- Stage 1: fp32 -> f16 conversion ----
__global__ __launch_bounds__(256) void cvt_f32_f16(const float* __restrict__ src,
                                                   _Float16* __restrict__ dst, int n) {
  int i = blockIdx.x * 256 + threadIdx.x;
  if (i < n) dst[i] = (_Float16)src[i];
}

// ---- Stage 2: QKV GEMM [4096,1024]x[1024,3072]^T + bias, scatter to Q/K/Vt ----
// Each wave: 32x64 output tile (2 A-frags x 4 B-frags = 8 WMMA per k-step).
// torch split order: K, Q, V along the 3C axis. Q pre-scaled by hd^-0.5 = 0.125.
__global__ __launch_bounds__(256) void qkv_gemm(const _Float16* __restrict__ X,
                                                const _Float16* __restrict__ W,
                                                const float* __restrict__ bias,
                                                _Float16* __restrict__ Qh,
                                                _Float16* __restrict__ Kh,
                                                _Float16* __restrict__ Vt) {
  const int lane = threadIdx.x & 31;
  const int wid  = (blockIdx.x * 256 + threadIdx.x) >> 5;   // global wave id
  const int mt = wid / 48;            // 128 M-tiles of 32
  const int ng = wid % 48;            // 48 groups of 4 N-tiles (3072 cols)
  const int m0 = mt * 32, n0 = ng * 64;

  const _Float16* Abase = X + (long)m0 * C_DIM;
  v8f c[2][4] = {};
  for (int k0 = 0; k0 < C_DIM; k0 += 32) {
    v16h a0 = load_a_f16(Abase + k0, C_DIM, lane);
    v16h a1 = load_a_f16(Abase + (long)16 * C_DIM + k0, C_DIM, lane);
    __builtin_prefetch(Abase + k0 + 64, 0, 3);
#pragma unroll
    for (int j = 0; j < 4; ++j) {
      v16h b = load_b_f16(W + (long)(n0 + j * 16) * C_DIM + k0, C_DIM, lane);
      c[0][j] = wmma_f32(a0, b, c[0][j]);
      c[1][j] = wmma_f32(a1, b, c[1][j]);
    }
  }
  const int col = lane & 15;
  const int rbase = (lane >> 4) << 3;
#pragma unroll
  for (int j = 0; j < 4; ++j) {
    const int n = n0 + j * 16 + col;
    const float bv = bias[n];
    const int split = n >> 10;          // 0:K 1:Q 2:V
    const int h = (n & 1023) >> 6;
    const int d = n & 63;
#pragma unroll
    for (int i = 0; i < 2; ++i) {
#pragma unroll
      for (int r = 0; r < 8; ++r) {
        const int m = m0 + i * 16 + rbase + r;
        const int bb = m >> 11;           // batch
        const int t  = m & 2047;          // token
        const int bh = (bb << 4) | h;
        const float v = c[i][j][r] + bv;
        if (split == 0)      Kh[((long)((bh << 11) | t)) * 64 + d] = (_Float16)v;
        else if (split == 1) Qh[((long)((bh << 11) | t)) * 64 + d] = (_Float16)(v * 0.125f);
        else                 Vt[((long)((bh << 6) | d)) * T_SEQ + t] = (_Float16)v;
      }
    }
  }
}

// ---- Stage 3: causal flash attention, one 16-row Q tile per wave ----
// 4 waves / block to keep static LDS < 64KB (double-buffered TDM panels).
__global__ __launch_bounds__(128) void attn_kernel(const _Float16* __restrict__ Qh,
                                                   const _Float16* __restrict__ Kh,
                                                   const _Float16* __restrict__ Vt,
                                                   _Float16* __restrict__ Ah) {
  __shared__ _Float16 pbuf[4][16 * 48];        // per-wave P staging (row stride 48)
#if HAVE_TDM
  __shared__ _Float16 vpanel[4][2][64 * 32];   // per-wave double-buffered V^T panels
#endif
  const int lane = threadIdx.x & 31;
  const int w    = threadIdx.x >> 5;
  const int tile = blockIdx.x * 4 + w;    // 4096 tiles total
  const int bh = tile >> 7;               // 0..31  (b*16+h)
  const int qt = tile & 127;              // q tile index
  const int col = lane & 15;
  const int rbase = (lane >> 4) << 3;

  const _Float16* Qbase = Qh + ((long)bh * T_SEQ + qt * 16) * 64;
  const v16h q0 = load_a_f16(Qbase, 64, lane);        // d = 0..31
  const v16h q1 = load_a_f16(Qbase + 32, 64, lane);   // d = 32..63

  v8f o0 = {}, o1 = {}, o2 = {}, o3 = {};
  float mrow[8], lrow[8];
#pragma unroll
  for (int r = 0; r < 8; ++r) { mrow[r] = NEG_BIG; lrow[r] = 0.f; }

  const _Float16* Kbh = Kh + (long)bh * T_SEQ * 64;
  const _Float16* Vbh = Vt + (long)bh * 64 * T_SEQ;
  _Float16* pb = &pbuf[w][0];
#if HAVE_TDM
  _Float16* vpan[2] = { &vpanel[w][0][0], &vpanel[w][1][0] };
  const unsigned vlds[2] = { (unsigned)(unsigned long)vpan[0],
                             (unsigned)(unsigned long)vpan[1] };
  // Preload first V^T panel [64 rows (d) x 32 keys] while Q-frags load.
  tdm_load_2d_f16(vlds[0], (unsigned long)(const void*)Vbh,
                  /*tile*/32, 64, /*tensor rem*/T_SEQ, 64, /*stride*/T_SEQ);
#endif

  for (int k0 = 0; k0 <= qt * 16; k0 += 32) {
    const int par = (k0 >> 5) & 1;
    const int knext = k0 + 32;
    const bool more = (knext <= qt * 16);
#if HAVE_TDM
    // DMA the NEXT panel into the other buffer; overlaps the full iteration.
    if (more)
      tdm_load_2d_f16(vlds[par ^ 1], (unsigned long)(const void*)(Vbh + knext),
                      32, 64, (unsigned)(T_SEQ - knext), 64, T_SEQ);
#endif
    // S = Q * K^T for keys [k0, k0+32): two 16x16 f32 tiles, K-dim = 64 (2 steps)
    v8f s0 = {}, s1 = {};
    s0 = wmma_f32(q0, load_b_f16(Kbh + (long)k0 * 64, 64, lane), s0);
    s0 = wmma_f32(q1, load_b_f16(Kbh + (long)k0 * 64 + 32, 64, lane), s0);
    s1 = wmma_f32(q0, load_b_f16(Kbh + (long)(k0 + 16) * 64, 64, lane), s1);
    s1 = wmma_f32(q1, load_b_f16(Kbh + (long)(k0 + 16) * 64 + 32, 64, lane), s1);

    if (k0 + 31 > qt * 16) {       // block straddles the causal diagonal
#pragma unroll
      for (int r = 0; r < 8; ++r) {
        const int qrow = qt * 16 + rbase + r;
        if (k0 + col > qrow)      s0[r] = NEG_BIG;
        if (k0 + 16 + col > qrow) s1[r] = NEG_BIG;
      }
    }

    // Online softmax. Row r of the tile lives in VGPR r across a 16-lane
    // group (lanes 0-15: rows 0-7, lanes 16-31: rows 8-15) -> xor-reduce.
    float p0[8], p1[8];
#pragma unroll
    for (int r = 0; r < 8; ++r) {
      float pm = fmaxf(s0[r], s1[r]);
      pm = fmaxf(pm, __shfl_xor(pm, 1, 32));
      pm = fmaxf(pm, __shfl_xor(pm, 2, 32));
      pm = fmaxf(pm, __shfl_xor(pm, 4, 32));
      pm = fmaxf(pm, __shfl_xor(pm, 8, 32));
      const float mnew  = fmaxf(mrow[r], pm);
      const float alpha = __expf(mrow[r] - mnew);
      p0[r] = __expf(s0[r] - mnew);
      p1[r] = __expf(s1[r] - mnew);
      float ps = p0[r] + p1[r];
      ps += __shfl_xor(ps, 1, 32);
      ps += __shfl_xor(ps, 2, 32);
      ps += __shfl_xor(ps, 4, 32);
      ps += __shfl_xor(ps, 8, 32);
      lrow[r] = lrow[r] * alpha + ps;
      mrow[r] = mnew;
      o0[r] *= alpha; o1[r] *= alpha; o2[r] *= alpha; o3[r] *= alpha;
    }

    // Relayout P (C/D layout) -> A fragment via LDS (same-wave DS is in-order).
#pragma unroll
    for (int r = 0; r < 8; ++r) {
      pb[(rbase + r) * 48 + col]      = (_Float16)p0[r];
      pb[(rbase + r) * 48 + 16 + col] = (_Float16)p1[r];
    }
    asm volatile("s_wait_dscnt 0x0" ::: "memory");
    const _Float16* prow = pb + (lane & 15) * 48 + ((lane >> 4) << 3);
    V16HU u;
    u.h[0] = *(const v8h*)(prow);
    u.h[1] = *(const v8h*)(prow + 16);
    const v16h pf = u.v;

    // O += P * V : B-frags from the TDM-staged LDS panel (contiguous ds reads).
#if HAVE_TDM
    // TDM ops of one wave complete in order: <=1 outstanding means the
    // current panel is resident (2 outstanding only right after 'more' issue).
    if (more) __builtin_amdgcn_s_wait_tensorcnt(1);
    else      __builtin_amdgcn_s_wait_tensorcnt(0);
    asm volatile("" ::: "memory");
    const _Float16* vp = vpan[par];
    o0 = wmma_f32(pf, load_b_f16(vp + 0 * 16 * 32, 32, lane), o0);
    o1 = wmma_f32(pf, load_b_f16(vp + 1 * 16 * 32, 32, lane), o1);
    o2 = wmma_f32(pf, load_b_f16(vp + 2 * 16 * 32, 32, lane), o2);
    o3 = wmma_f32(pf, load_b_f16(vp + 3 * 16 * 32, 32, lane), o3);
#else
    o0 = wmma_f32(pf, load_b_f16(Vbh + (long)(0 * 16) * T_SEQ + k0, T_SEQ, lane), o0);
    o1 = wmma_f32(pf, load_b_f16(Vbh + (long)(1 * 16) * T_SEQ + k0, T_SEQ, lane), o1);
    o2 = wmma_f32(pf, load_b_f16(Vbh + (long)(2 * 16) * T_SEQ + k0, T_SEQ, lane), o2);
    o3 = wmma_f32(pf, load_b_f16(Vbh + (long)(3 * 16) * T_SEQ + k0, T_SEQ, lane), o3);
#endif
  }

  // Normalize and write [B,T,C] f16 for the output projection.
  const int b = bh >> 4, h = bh & 15;
#pragma unroll
  for (int r = 0; r < 8; ++r) {
    const int t = qt * 16 + rbase + r;
    _Float16* dst = Ah + ((long)(b * T_SEQ + t)) * C_DIM + h * 64;
    const float inv = 1.0f / lrow[r];
    dst[0 * 16 + col] = (_Float16)(o0[r] * inv);
    dst[1 * 16 + col] = (_Float16)(o1[r] * inv);
    dst[2 * 16 + col] = (_Float16)(o2[r] * inv);
    dst[3 * 16 + col] = (_Float16)(o3[r] * inv);
  }
}

// ---- Stage 4: output projection [4096,1024]x[1024,1024]^T + bias -> f32 ----
// Each wave: 32x64 output tile.
__global__ __launch_bounds__(256) void oproj_gemm(const _Float16* __restrict__ Ah,
                                                  const _Float16* __restrict__ Wo,
                                                  const float* __restrict__ bo,
                                                  float* __restrict__ out) {
  const int lane = threadIdx.x & 31;
  const int wid  = (blockIdx.x * 256 + threadIdx.x) >> 5;
  const int mt = wid >> 4;            // 128 M-tiles of 32
  const int ng = wid & 15;            // 16 groups of 4 N-tiles
  const int m0 = mt * 32, n0 = ng * 64;

  const _Float16* Abase = Ah + (long)m0 * C_DIM;
  v8f c[2][4] = {};
  for (int k0 = 0; k0 < C_DIM; k0 += 32) {
    v16h a0 = load_a_f16(Abase + k0, C_DIM, lane);
    v16h a1 = load_a_f16(Abase + (long)16 * C_DIM + k0, C_DIM, lane);
    __builtin_prefetch(Abase + k0 + 64, 0, 3);
#pragma unroll
    for (int j = 0; j < 4; ++j) {
      v16h b = load_b_f16(Wo + (long)(n0 + j * 16) * C_DIM + k0, C_DIM, lane);
      c[0][j] = wmma_f32(a0, b, c[0][j]);
      c[1][j] = wmma_f32(a1, b, c[1][j]);
    }
  }
  const int col = lane & 15;
  const int rbase = (lane >> 4) << 3;
#pragma unroll
  for (int j = 0; j < 4; ++j) {
    const int n = n0 + j * 16 + col;
    const float bias = bo[n];
#pragma unroll
    for (int i = 0; i < 2; ++i)
#pragma unroll
      for (int r = 0; r < 8; ++r)
        out[(long)(m0 + i * 16 + rbase + r) * C_DIM + n] = c[i][j][r] + bias;
  }
}

extern "C" void kernel_launch(void* const* d_in, const int* in_sizes, int n_in,
                              void* d_out, int out_size, void* d_ws, size_t ws_size,
                              hipStream_t stream) {
  const float* x     = (const float*)d_in[0];   // [2,2048,1024]
  const float* Wqkv  = (const float*)d_in[1];   // [3072,1024]
  const float* bqkv  = (const float*)d_in[2];   // [3072]
  const float* Wo    = (const float*)d_in[3];   // [1024,1024]
  const float* bo    = (const float*)d_in[4];   // [1024]
  float* out = (float*)d_out;                   // [2,2048,1024] f32

  char* ws = (char*)d_ws;
  size_t off = 0;
  auto alloc = [&](size_t bytes) -> void* {
    void* p = ws + off;
    off = (off + bytes + 255) & ~(size_t)255;
    return p;
  };
  _Float16* xh    = (_Float16*)alloc((size_t)4096 * 1024 * 2);    // x f16
  _Float16* wqkvh = (_Float16*)alloc((size_t)3072 * 1024 * 2);    // W_qkv f16
  _Float16* woh   = (_Float16*)alloc((size_t)1024 * 1024 * 2);    // W_o f16
  _Float16* qh    = (_Float16*)alloc((size_t)32 * 2048 * 64 * 2); // Q [bh][t][d] (pre-scaled)
  _Float16* kh    = (_Float16*)alloc((size_t)32 * 2048 * 64 * 2); // K [bh][t][d]
  _Float16* vth   = (_Float16*)alloc((size_t)32 * 64 * 2048 * 2); // V^T [bh][d][t]
  _Float16* ah    = (_Float16*)alloc((size_t)4096 * 1024 * 2);    // attn out [B,T,C]

  cvt_f32_f16<<<16384, 256, 0, stream>>>(x, xh, 4096 * 1024);
  cvt_f32_f16<<<12288, 256, 0, stream>>>(Wqkv, wqkvh, 3072 * 1024);
  cvt_f32_f16<<<4096, 256, 0, stream>>>(Wo, woh, 1024 * 1024);

  // 128 M-tiles x 48 groups = 6144 waves = 768 blocks of 8 waves
  qkv_gemm<<<768, 256, 0, stream>>>(xh, wqkvh, bqkv, qh, kh, vth);

  // 32 (b,h) x 128 q-tiles = 4096 waves = 1024 blocks of 4 waves
  attn_kernel<<<1024, 128, 0, stream>>>(qh, kh, vth, ah);

  // 128 M-tiles x 16 groups = 2048 waves = 256 blocks
  oproj_gemm<<<256, 256, 0, stream>>>(ah, woh, bo, out);
}
